// Pedal_26482768347620
// MI455X (gfx1250) — matrix operand order repaired and checked by reading, not compiled, not run on hardware
//
#include <hip/hip_runtime.h>
#include <hip/hip_bf16.h>
#include <math.h>

#define P_ 6
#define B_ 256
#define D_ 256
#define N_ 32768
#define KTOP 10
#define NCHUNKS 32
#define CHUNK 1024          // N_/NCHUNKS columns per workgroup
#define CTILE 32            // columns per inner iteration
#define BLD 264             // LDS B stride in halves (pad vs 256 to dodge bank conflicts)
#define DLD 33              // LDS dist stride in floats (pad vs 32)
#define THRS 0.7f
#define NEGV -1e30f

typedef __attribute__((ext_vector_type(16))) _Float16 v16h;
typedef __attribute__((ext_vector_type(8)))  float    v8f;
typedef __attribute__((ext_vector_type(4)))  float    v4f;

__device__ __forceinline__ v8f wmma16(v16h a, v16h b, v8f c) {
  // D = A(16x32 f16) * B(32x16 f16) + C(16x16 f32)
  return __builtin_amdgcn_wmma_f32_16x16x32_f16(false, a, false, b, (short)0, c,
                                                false, false);
}

// A fragment (16x32 f16, row-major source, row stride ld halves).
// lanes 0-15: row=lane, K {0..7,16..23}; lanes 16-31: row=lane-16, K {8..15,24..31}
__device__ __forceinline__ v16h frag_a(const _Float16* base, int ld, int k0, int lane) {
  int r  = lane & 15;
  int ks = (lane >> 4) & 1;
  const _Float16* p = base + (size_t)r * ld + k0 + ks * 8;
  union { v16h h; v4f f[2]; } u;
  u.f[0] = *reinterpret_cast<const v4f*>(p);
  u.f[1] = *reinterpret_cast<const v4f*>(p + 16);
  return u.h;
}

// B fragment (32x16 f16). Column n stored contiguously along K (stride ld halves).
// lanes 0-15: col=lane, K {0..7,16..23}; lanes 16-31: col=lane-16, K {8..15,24..31}
__device__ __forceinline__ v16h frag_b(const _Float16* base, int ld, int k0, int lane) {
  int c  = lane & 15;
  int ks = (lane >> 4) & 1;
  const _Float16* p = base + (size_t)c * ld + k0 + ks * 8;
  union { v16h h; v4f f[2]; } u;
  u.f[0] = *reinterpret_cast<const v4f*>(p);
  u.f[1] = *reinterpret_cast<const v4f*>(p + 16);
  return u.h;
}

// sorted ascending insertion; strict < keeps earlier index on ties (stable argsort),
// NaN never inserted (matches argsort pushing NaN last)
__device__ __forceinline__ void top_insert(float (&td)[KTOP], int (&ti)[KTOP],
                                           float d, int idx) {
  if (!(d < td[KTOP - 1])) return;
  float cd = d; int ci = idx;
#pragma unroll
  for (int k = 0; k < KTOP; ++k) {
    if (cd < td[k]) {
      float tf = td[k]; td[k] = cd; cd = tf;
      int   tt = ti[k]; ti[k] = ci; ci = tt;
    }
  }
}

// ---------------- K0: zero the accumulators in workspace ----------------
__global__ void k0_init(float* expsumW, float* perP, float* alignL) {
  int t = threadIdx.x;
  for (int i = t; i < P_ * B_; i += 256) expsumW[i] = 0.f;
  if (t < P_) perP[t] = 0.f;
  if (t == 0) alignL[0] = 0.f;
}

// ---------------- K1: normalize, aligned features, row norms ----------------
__global__ void k1_align(const float* __restrict__ feature, const float* __restrict__ text,
                         _Float16* alignedH, _Float16* imgH, _Float16* txH,
                         float* fnorm2) {
  int bid = blockIdx.x;            // p*B + b
  int p = bid / B_, b = bid % B_;
  int t = threadIdx.x;             // d
  __shared__ float s1[256], s2[256];
  float f = feature[((size_t)p * B_ + b) * D_ + t];
  float x = text[((size_t)b * P_ + p) * D_ + t];
  s1[t] = f * f; s2[t] = x * x;
  __syncthreads();
  for (int o = 128; o > 0; o >>= 1) {
    if (t < o) { s1[t] += s1[t + o]; s2[t] += s2[t + o]; }
    __syncthreads();
  }
  float nf = fmaxf(sqrtf(s1[0]), 1e-12f);
  float nt = fmaxf(sqrtf(s2[0]), 1e-12f);
  __syncthreads();
  float img = f / nf, tx = x / nt;
  float al = 0.9f * img + 0.1f * tx;   // img + 0.1*(tx - stopgrad(img))
  size_t oo = ((size_t)p * B_ + b) * D_ + t;
  alignedH[oo] = (_Float16)al;
  imgH[oo] = (_Float16)img;
  txH[oo]  = (_Float16)tx;
  s1[t] = al * al;
  __syncthreads();
  for (int o = 128; o > 0; o >>= 1) {
    if (t < o) s1[t] += s1[t + o];
    __syncthreads();
  }
  if (t == 0) fnorm2[p * B_ + b] = s1[0];
}

// ---------------- K2a: gram matrices img@img^T and tx@tx^T via WMMA ----------------
__global__ void k2a_gram(const _Float16* __restrict__ imgH, const _Float16* __restrict__ txH,
                         float* simImg, float* simTxt) {
  int bid   = blockIdx.x;           // 48 = P * 2 * 4
  int p     = bid >> 3;
  int which = (bid >> 2) & 1;
  int rowblk = bid & 3;
  const _Float16* src = (which == 0 ? imgH : txH) + (size_t)p * B_ * D_;
  float* dst = (which == 0 ? simImg : simTxt) + (size_t)p * B_ * B_;
  int t = threadIdx.x, lane = t & 31, wv = t >> 5;
  int r0 = rowblk * 64 + (wv & 3) * 16;
  int cb = (wv >> 2) * 128;
  const v8f vz = {0, 0, 0, 0, 0, 0, 0, 0};
  v8f acc[8];
#pragma unroll
  for (int i = 0; i < 8; ++i) acc[i] = vz;
#pragma unroll
  for (int ks = 0; ks < 8; ++ks) {
    v16h a = frag_a(src + (size_t)r0 * D_, D_, ks * 32, lane);
#pragma unroll
    for (int ct = 0; ct < 8; ++ct) {
      // B = src^T: column n of B is row n of src -> contiguous along K
      v16h b = frag_b(src + (size_t)(cb + ct * 16) * D_, D_, ks * 32, lane);
      acc[ct] = wmma16(a, b, acc[ct]);
    }
  }
#pragma unroll
  for (int ct = 0; ct < 8; ++ct)
#pragma unroll
    for (int v = 0; v < 8; ++v) {
      int row = r0 + v + ((lane >> 4) << 3);
      int col = cb + ct * 16 + (lane & 15);
      dst[(size_t)row * B_ + col] = 2.0f * acc[ct][v];   // 1/TEMP
    }
}

// ---------------- K2b: masked log-softmax symmetric KL (align_loss) ----------------
__global__ void k2b_alignloss(const float* __restrict__ simImg,
                              const float* __restrict__ simTxt, float* alignL) {
  int p = blockIdx.x;
  int i = threadIdx.x;                     // row
  __shared__ int validL[B_];
  __shared__ float r1[B_], r2[B_];
  __shared__ int nv;
  const float* SI = simImg + ((size_t)p * B_ + i) * B_;
  const float* ST = simTxt + ((size_t)p * B_ + i) * B_;
  int valid = 0;
  for (int j = 0; j < B_; ++j)
    if (SI[j] > THRS && ST[j] > THRS) { valid = 1; break; }
  validL[i] = valid;
  if (i == 0) nv = 0;
  __syncthreads();
  if (valid) atomicAdd(&nv, 1);
  __syncthreads();
  float mI = NEGV, mT = NEGV;
  for (int j = 0; j < B_; ++j)
    if (validL[j]) { mI = fmaxf(mI, SI[j]); mT = fmaxf(mT, ST[j]); }
  float sI = 0.f, sT = 0.f;
  for (int j = 0; j < B_; ++j)
    if (validL[j]) { sI += __expf(SI[j] - mI); sT += __expf(ST[j] - mT); }
  float lI = mI + logf(sI), lT = mT + logf(sT);
  float kl1 = 0.f, kl2 = 0.f;
  if (valid) {
    for (int j = 0; j < B_; ++j)
      if (validL[j]) {
        float a = SI[j] - lI;              // img_logp
        float c = ST[j] - lT;              // txt_logp
        kl1 += __expf(c) * (c - a);
        kl2 += __expf(a) * (a - c);
      }
  }
  r1[i] = kl1; r2[i] = kl2;
  __syncthreads();
  for (int o = 128; o > 0; o >>= 1) {
    if (i < o) { r1[i] += r1[i + o]; r2[i] += r2[i + o]; }
    __syncthreads();
  }
  if (i == 0 && nv > 0) {
    float lk = 0.5f * (r1[0] + r2[0]) / (float)nv;
    atomicAdd(alignL, lk);
  }
}

// ---------------- K3: fused dist GEMM + per-row top-K + expsum ----------------
// Software-pipelined: tile i+1's f32 center columns are loaded into registers
// while tile i is being multiplied/scanned, so HBM latency overlaps WMMA.
__global__ void __launch_bounds__(256)
k3_dist(const _Float16* __restrict__ alignedH, const float* __restrict__ centers,
        const float* __restrict__ fnorm2, const int* __restrict__ position,
        float* candD, int* candI, float* expsumW) {
  int bid = blockIdx.x;
  int p = bid / NCHUNKS;
  int chunk = bid % NCHUNKS;
  int chunkbase = chunk * CHUNK;
  int t = threadIdx.x, lane = t & 31, wv = t >> 5;

  __shared__ _Float16 ldsB[CTILE * BLD];     // 16.9 KB f16 center tile (K-major per col)
  __shared__ float ldsDist[B_ * DLD];        // 33.8 KB dot-product spill
  __shared__ float ldsCn[CTILE];             // exact f32 column norms

  // Preload this wave's A fragments for the full K=256 (rows fixed all kernel)
  v16h A0[8], A1[8];
  const _Float16* ab = alignedH + ((size_t)p * B_ + wv * 32) * D_;
#pragma unroll
  for (int ks = 0; ks < 8; ++ks) {
    A0[ks] = frag_a(ab, D_, ks * 32, lane);
    A1[ks] = frag_a(ab + 16 * D_, D_, ks * 32, lane);
  }

  int r = t;                                 // this thread's row for scanning
  float fn = fnorm2[p * B_ + r];
  int pos = position[r];
  float esum = 0.f;
  float td[KTOP]; int ti[KTOP];
#pragma unroll
  for (int k = 0; k < KTOP; ++k) { td[k] = 1e30f; ti[k] = -1; }

  int colS = t >> 3;                         // staging: 8 threads per column
  int sub = t & 7;                           // each covers 32 K values
  const float* cbase = centers + ((size_t)p * N_ + chunkbase + colS) * D_ + sub * 32;

  // pipeline prologue: tile 0 -> registers
  float2 rv[16];
#pragma unroll
  for (int q = 0; q < 16; ++q)
    rv[q] = *reinterpret_cast<const float2*>(cbase + 2 * q);

  for (int it = 0; it < CHUNK / CTILE; ++it) {
    if (t < CTILE) ldsCn[t] = 0.f;
    __syncthreads();
    { // convert register tile (f32 -> f16) into LDS, accumulate exact ||c||^2
      float part = 0.f;
#pragma unroll
      for (int q = 0; q < 16; ++q) {
        float2 v = rv[q];
        part += v.x * v.x + v.y * v.y;
        unsigned short h0 = __builtin_bit_cast(unsigned short, (_Float16)v.x);
        unsigned short h1 = __builtin_bit_cast(unsigned short, (_Float16)v.y);
        unsigned int pk = ((unsigned int)h1 << 16) | h0;
        *reinterpret_cast<unsigned int*>(&ldsB[colS * BLD + sub * 32 + 2 * q]) = pk;
      }
      atomicAdd(&ldsCn[colS], part);         // ds_add_f32
    }
    // issue next tile's loads now; they complete during WMMA + scan below
    if (it + 1 < CHUNK / CTILE) {
      const float* cp = cbase + (size_t)(it + 1) * CTILE * D_;
      __builtin_prefetch(cp + (size_t)CTILE * D_, 0, 1);   // warm L2 for tile i+2
#pragma unroll
      for (int q = 0; q < 16; ++q)
        rv[q] = *reinterpret_cast<const float2*>(cp + 2 * q);
    }
    __syncthreads();
    // 256x32 dot block: 8 waves x (2 row-tiles x 2 col-tiles) x 8 K-steps
    const v8f vz = {0, 0, 0, 0, 0, 0, 0, 0};
    v8f a00 = vz, a01 = vz, a10 = vz, a11 = vz;
#pragma unroll
    for (int ks = 0; ks < 8; ++ks) {
      v16h b0 = frag_b(&ldsB[0], BLD, ks * 32, lane);
      v16h b1 = frag_b(&ldsB[16 * BLD], BLD, ks * 32, lane);
      a00 = wmma16(A0[ks], b0, a00);
      a01 = wmma16(A0[ks], b1, a01);
      a10 = wmma16(A1[ks], b0, a10);
      a11 = wmma16(A1[ks], b1, a11);
    }
    { // spill dot tile to LDS (row-major, padded)
      int rb = wv * 32 + ((lane >> 4) << 3);
      int cl = lane & 15;
#pragma unroll
      for (int v = 0; v < 8; ++v) {
        ldsDist[(rb + v) * DLD + cl]           = a00[v];
        ldsDist[(rb + v) * DLD + 16 + cl]      = a01[v];
        ldsDist[(rb + 16 + v) * DLD + cl]      = a10[v];
        ldsDist[(rb + 16 + v) * DLD + 16 + cl] = a11[v];
      }
    }
    __syncthreads();
    // per-row scan: dist = ||f||^2 + ||c||^2 - 2 dot; own position skipped (== inf)
    int nb = chunkbase + it * CTILE;
    for (int j = 0; j < CTILE; ++j) {
      int n = nb + j;
      float dd = fn + ldsCn[j] - 2.0f * ldsDist[r * DLD + j];
      if (n != pos) {
        esum += __expf(-10.0f * dd);
        top_insert(td, ti, dd, n);
      }
    }
    __syncthreads();
  }
  size_t cb = (((size_t)p * NCHUNKS + chunk) * B_ + r) * KTOP;
#pragma unroll
  for (int k = 0; k < KTOP; ++k) { candD[cb + k] = td[k]; candI[cb + k] = ti[k]; }
  atomicAdd(&expsumW[p * B_ + r], esum);     // global_atomic_add_f32
}

// ---------------- K4: merge chunk top-Ks, gather vids, per-p loss ----------------
__global__ void k4_merge(const float* __restrict__ candD, const int* __restrict__ candI,
                         const float* __restrict__ expsumW,
                         const int* __restrict__ position, const int* __restrict__ vid_mem,
                         float* outp, float* perP) {
  int tid = blockIdx.x * 256 + threadIdx.x;  // 0..P*B-1
  int p = tid / B_, b = tid % B_;
  float td[KTOP]; int ti[KTOP];
#pragma unroll
  for (int k = 0; k < KTOP; ++k) { td[k] = 1e30f; ti[k] = -1; }
  for (int c = 0; c < NCHUNKS; ++c) {        // chunk order == ascending n (stable)
    size_t cb = (((size_t)p * NCHUNKS + c) * B_ + b) * KTOP;
    for (int k = 0; k < KTOP; ++k) {
      int idx = candI[cb + k];
      if (idx < 0) break;                    // candidates sorted; -1 means empty
      float d = candD[cb + k];
      if (!(d < td[KTOP - 1])) break;
      top_insert(td, ti, d, idx);
    }
  }
  int pos = position[b];
  float sx = 0.f;
#pragma unroll
  for (int k = 0; k < KTOP; ++k) {
    sx += __expf(-10.0f * td[k]);
    int idx = ti[k];
    int f = idx - (idx > pos ? 1 : 0);
    f = f < 0 ? 0 : (f > N_ - 2 ? N_ - 2 : f);
    outp[1 + ((size_t)p * B_ + b) * KTOP + k] = (float)vid_mem[f];
  }
  float y = logf(expsumW[p * B_ + b]);
  float x = logf(sx);
  atomicAdd(&perP[p], (y - x) * (1.0f / (float)B_));
}

// ---------------- K5: final scalar loss ----------------
__global__ void k5_final(const float* perP, const float* alignL, float* outp) {
  if (threadIdx.x == 0 && blockIdx.x == 0) {
    float s = 0.f;
    for (int p = 0; p < P_; ++p) { float l = perP[p]; s += (l == l) ? l : 0.f; }
    outp[0] = s / (float)P_ + 0.5f * alignL[0];
  }
}

extern "C" void kernel_launch(void* const* d_in, const int* in_sizes, int n_in,
                              void* d_out, int out_size, void* d_ws, size_t ws_size,
                              hipStream_t stream) {
  (void)in_sizes; (void)n_in; (void)out_size; (void)ws_size;
  const float* feature = (const float*)d_in[0];
  const float* text    = (const float*)d_in[1];
  const float* centers = (const float*)d_in[2];
  const int*   position = (const int*)d_in[3];
  const int*   vid_mem  = (const int*)d_in[4];
  float* outp = (float*)d_out;

  char* w = (char*)d_ws;
  size_t off = 0;
  auto take = [&](size_t bytes) -> void* {
    void* r = w + off;
    off += (bytes + 255) & ~(size_t)255;
    return r;
  };
  _Float16* alignedH = (_Float16*)take((size_t)P_ * B_ * D_ * 2);
  _Float16* imgH     = (_Float16*)take((size_t)P_ * B_ * D_ * 2);
  _Float16* txH      = (_Float16*)take((size_t)P_ * B_ * D_ * 2);
  float* fnorm2  = (float*)take((size_t)P_ * B_ * 4);
  float* simImg  = (float*)take((size_t)P_ * B_ * B_ * 4);
  float* simTxt  = (float*)take((size_t)P_ * B_ * B_ * 4);
  float* expsumW = (float*)take((size_t)P_ * B_ * 4);
  float* perP    = (float*)take(256);
  float* alignL  = (float*)take(256);
  float* candD   = (float*)take((size_t)P_ * NCHUNKS * B_ * KTOP * 4);
  int*   candI   = (int*)take((size_t)P_ * NCHUNKS * B_ * KTOP * 4);

  k0_init<<<1, 256, 0, stream>>>(expsumW, perP, alignL);
  k1_align<<<P_ * B_, 256, 0, stream>>>(feature, text, alignedH, imgH, txH, fnorm2);
  k2a_gram<<<P_ * 2 * 4, 256, 0, stream>>>(imgH, txH, simImg, simTxt);
  k2b_alignloss<<<P_, 256, 0, stream>>>(simImg, simTxt, alignL);
  k3_dist<<<P_ * NCHUNKS, 256, 0, stream>>>(alignedH, centers, fnorm2, position,
                                            candD, candI, expsumW);
  k4_merge<<<P_ * B_ / 256, 256, 0, stream>>>(candD, candI, expsumW, position, vid_mem,
                                              outp, perP);
  k5_final<<<1, 64, 0, stream>>>(perP, alignL, outp);
}
